// PersonalMHA_1185410974346
// MI455X (gfx1250) — matrix operand order repaired and checked by reading, not compiled, not run on hardware
//
#include <hip/hip_runtime.h>

// ---------------- problem constants (from reference) ----------------
#define BQ     4
#define S_LEN  2048
#define D_HID  512
#define NH     8
#define DH     64
#define DP1    513      // D+1 input width
#define KPAD   544      // 513 padded to multiple of 32 (WMMA K-step)
#define VROWS  80       // dh+1 = 65 padded to 5 * 16 N-tiles
#define ROWS   (BQ * S_LEN)  // 8192 flattened (b,s) rows

// LDS row padding (elements) to rotate across the 64 4B banks
#define KLROW  72       // 32B-fragment reads: stride 144B -> 36 words mod 64
#define VLROW  40       // stride 80B -> 20 words mod 64
#define PLROW  40

typedef __attribute__((ext_vector_type(16))) __bf16        v16bf;
typedef __attribute__((ext_vector_type(8)))  float         v8f;
typedef __attribute__((ext_vector_type(4)))  unsigned int  uint4v;

struct bfpair { uint4v a, b; };

// Build a v16bf A/B fragment from two contiguous 8-element (16B) bf16 runs.
__device__ __forceinline__ v16bf load_frag2(const unsigned short* p0,
                                            const unsigned short* p1) {
  bfpair t;
  t.a = *(const uint4v*)p0;
  t.b = *(const uint4v*)p1;
  return __builtin_bit_cast(v16bf, t);
}

__device__ __forceinline__ unsigned short f2bf(float f) {
  unsigned int u = __builtin_bit_cast(unsigned int, f);
  unsigned int r = u + 0x7FFFu + ((u >> 16) & 1u);
  return (unsigned short)(r >> 16);
}

__device__ __forceinline__ v8f wmma_bf16(v16bf a, v16bf b, v8f c) {
  return __builtin_amdgcn_wmma_f32_16x16x32_bf16(
      /*neg_a=*/false, a, /*neg_b=*/false, b,
      /*c_mod=*/(short)0, c, /*reuse_a=*/false, /*reuse_b=*/false);
}

// Async global->LDS copy, 16B per enabled lane (VGLOBAL encoding: VDST holds
// the per-lane LDS byte offset, VADDR the 64-bit global address). ASYNCcnt.
__device__ __forceinline__ void async_copy16(unsigned lds_off,
                                             const unsigned short* g) {
  asm volatile("global_load_async_to_lds_b128 %0, %1, off"
               :: "v"(lds_off), "v"((unsigned long long)(uintptr_t)g)
               : "memory");
}
__device__ __forceinline__ void wait_async0() {
  asm volatile("s_wait_asynccnt 0x0" ::: "memory");
}

// Row reductions: in the CDNA5 16x16 f32 C/D layout a matrix row occupies one
// VGPR across a 16-lane half-wave, so xor-shuffles with masks 1,2,4,8 reduce
// exactly one row (bit 4 of the lane id is never touched).
__device__ __forceinline__ float redmax16(float v) {
  #pragma unroll
  for (int m = 1; m < 16; m <<= 1) v = fmaxf(v, __shfl_xor(v, m, 32));
  return v;
}
__device__ __forceinline__ float redsum16(float v) {
  #pragma unroll
  for (int m = 1; m < 16; m <<= 1) v += __shfl_xor(v, m, 32);
  return v;
}

// ---------------- pack f32 -> bf16 with zero K-padding ----------------
__global__ __launch_bounds__(256)
void k_pack(const float* __restrict__ src, unsigned short* __restrict__ dst,
            int rows, int incols, int outcols) {
  int idx = blockIdx.x * blockDim.x + threadIdx.x;
  if (idx >= rows * outcols) return;
  int r = idx / outcols, c = idx - r * outcols;
  float v = (c < incols) ? src[r * incols + c] : 0.0f;
  dst[idx] = f2bf(v);
}

__global__ __launch_bounds__(256)
void k_zero(unsigned short* __restrict__ p, int n) {
  int idx = blockIdx.x * blockDim.x + threadIdx.x;
  if (idx < n) p[idx] = 0;
}

// ---------------- QKV projection + hyperboloid lift (WMMA) -----------------
// Each wave computes one 16-row x 64-col (one head) output tile:
// s = X(16xKPAD) @ W^T, 17 K-steps x 4 N-tiles = 68 v_wmma_f32_16x16x32_bf16.
// Then t = sqrt(1 + ||s||^2) per row via 16-lane reductions.
// mode 0/1: write row-major bf16 space part (Qs/Ks) + f32 time (q0/k0).
// mode 2  : write V transposed [b,h,dim,seq] bf16, dim0 = time, 1..64 = space.
__global__ __launch_bounds__(256)
void k_project(const unsigned short* __restrict__ Xbf,
               const unsigned short* __restrict__ Wbf,
               const float* __restrict__ bias,
               unsigned short* __restrict__ outS,
               float* __restrict__ outT,
               unsigned short* __restrict__ outVt,
               int mode) {
  const int lane = threadIdx.x & 31;
  const int wave = threadIdx.x >> 5;
  const int h    = lane >> 4;        // half-wave
  const int lr   = lane & 15;
  const int rt   = blockIdx.x * 8 + wave;   // 16-row tile id (0..511)
  const int head = blockIdx.y;
  const int row0 = rt * 16;
  const int dbase = head * DH;

  v8f acc[4] = {};
  const unsigned short* arow = Xbf + (size_t)(row0 + lr) * KPAD;
  for (int ks = 0; ks < KPAD / 32; ++ks) {
    const int k0 = ks * 32;
    // A fragment (16x32 bf16): lane holds row lr, contiguous K = 16h..16h+15
    v16bf a = load_frag2(arow + k0 + 16 * h, arow + k0 + 16 * h + 8);
    #pragma unroll
    for (int t = 0; t < 4; ++t) {
      // B fragment (32x16): lane holds col n, contiguous K = 16h..16h+15
      const int n = dbase + t * 16 + lr;
      const unsigned short* brow = Wbf + (size_t)n * KPAD + k0 + 16 * h;
      v16bf b = load_frag2(brow, brow + 8);
      acc[t] = wmma_bf16(a, b, acc[t]);
    }
  }

  float bn[4];
  #pragma unroll
  for (int t = 0; t < 4; ++t) bn[t] = bias[dbase + t * 16 + lr];

  float sv[4][8], tval[8];
  #pragma unroll
  for (int r = 0; r < 8; ++r) {
    float part = 0.0f;
    #pragma unroll
    for (int t = 0; t < 4; ++t) {
      float v = acc[t][r] + bn[t];
      sv[t][r] = v;
      part += v * v;
    }
    tval[r] = sqrtf(1.0f + redsum16(part));   // c = 1
  }

  #pragma unroll
  for (int r = 0; r < 8; ++r) {
    const int m  = row0 + h * 8 + r;          // global row (b*S + s)
    const int bb = m >> 11;
    const int ss = m & (S_LEN - 1);
    const size_t bh = (size_t)(bb * NH + head);
    if (mode < 2) {
      const size_t base = (bh * S_LEN + ss) * DH;
      #pragma unroll
      for (int t = 0; t < 4; ++t)
        outS[base + t * 16 + lr] = f2bf(sv[t][r]);
      if (lr == 0) outT[bh * S_LEN + ss] = tval[r];
    } else {
      const size_t vbase = bh * (size_t)VROWS * S_LEN;
      #pragma unroll
      for (int t = 0; t < 4; ++t)
        outVt[vbase + (size_t)(1 + t * 16 + lr) * S_LEN + ss] = f2bf(sv[t][r]);
      if (lr == 0) outVt[vbase + ss] = f2bf(tval[r]);
    }
  }
}

// ---------------- flash attention with Lorentzian scores -----------------
// One wave owns a 16-query tile; 8 waves/block share K/V tiles staged in LDS
// by double-buffered GLOBAL_LOAD_ASYNC_TO_LDS (copies for chunk i+1 issued
// before computing chunk i; drained with s_wait_asynccnt + barrier).
// Per 32-key chunk: 4 WMMA (scores) + rank-1 (-q0*k0) correction, online
// softmax, P->LDS (bf16 C->A transpose), 5 WMMA (P @ V over 80 padded dims).
// Ends with Lorentz midpoint renorm, writes mid[B,S,H,65] f32.
__global__ __launch_bounds__(256)
void k_attn(const unsigned short* __restrict__ Qs, const float* __restrict__ q0,
            const unsigned short* __restrict__ Ks, const float* __restrict__ k0,
            const unsigned short* __restrict__ Vt,
            const unsigned char* __restrict__ mask,
            float* __restrict__ mid) {
  __shared__ alignas(32) unsigned short Kl[2][32 * KLROW];   // 9.2 KB
  __shared__ alignas(32) unsigned short Vl[2][VROWS * VLROW];// 12.8 KB
  __shared__ alignas(32) unsigned short Plds[8][16 * PLROW]; // 10.2 KB

  const int lane = threadIdx.x & 31;
  const int wave = threadIdx.x >> 5;
  const int h    = lane >> 4;
  const int lr   = lane & 15;
  const int head = blockIdx.y;
  const int bb   = blockIdx.z;
  const int qt   = blockIdx.x * 8 + wave;
  const int m0   = qt * 16;

  const size_t bh = (size_t)(bb * NH + head);
  const unsigned short* Qb  = Qs + bh * S_LEN * DH;
  const unsigned short* Kb  = Ks + bh * S_LEN * DH;
  const unsigned short* Vb  = Vt + bh * (size_t)VROWS * S_LEN;
  const float*          q0b = q0 + bh * S_LEN;
  const float*          k0b = k0 + bh * S_LEN;
  const unsigned char*  mb  = mask + (size_t)bb * S_LEN * S_LEN;

  // Stage one 32-key chunk: K = 32 keys x 128B (256 x 16B), V = 80 dims x 64B
  // (320 x 16B). 576 transfers spread over 256 threads, 16B per lane.
  auto stage = [&](int buf, int kv) {
    for (int i = threadIdx.x; i < 576; i += 256) {
      if (i < 256) {
        const int key = i >> 3, part = i & 7;
        async_copy16((unsigned)(uintptr_t)&Kl[buf][key * KLROW + part * 8],
                     Kb + (size_t)(kv + key) * DH + part * 8);
      } else {
        const int j = i - 256, dim = j >> 2, part = j & 3;
        async_copy16((unsigned)(uintptr_t)&Vl[buf][dim * VLROW + part * 8],
                     Vb + (size_t)dim * S_LEN + kv + part * 8);
      }
    }
  };

  // Q A-fragments for both K-steps (dims 0..31 and 32..63), reused all chunks
  const unsigned short* qrow = Qb + (size_t)(m0 + lr) * DH;
  const v16bf qa0 = load_frag2(qrow + 16 * h,      qrow + 8 + 16 * h);
  const v16bf qa1 = load_frag2(qrow + 32 + 16 * h, qrow + 40 + 16 * h);
  float q0r[8];
  #pragma unroll
  for (int r = 0; r < 8; ++r) q0r[r] = q0b[m0 + h * 8 + r];

  float mrow[8], lrow[8];
  #pragma unroll
  for (int r = 0; r < 8; ++r) { mrow[r] = -1e30f; lrow[r] = 0.0f; }
  v8f acc[5] = {};

  const float scale = 0.125f;   // 1/sqrt(dh) = 1/8
  unsigned short* Pl = &Plds[wave][0];

  stage(0, 0);
  wait_async0();
  __syncthreads();

  for (int kv = 0; kv < S_LEN; kv += 32) {
    const int p = (kv >> 5) & 1;
    if (kv + 32 < S_LEN) stage(p ^ 1, kv + 32);   // overlap copy with compute

    // ---- score tiles (2 N-tiles of 16 keys) from LDS-resident K ----
    v8f s0 = {}, s1 = {};
    {
      const unsigned short* kr0 = &Kl[p][(lr)      * KLROW + 16 * h];
      const unsigned short* kr1 = &Kl[p][(16 + lr) * KLROW + 16 * h];
      s0 = wmma_bf16(qa0, load_frag2(kr0,      kr0 + 8),  s0);
      s0 = wmma_bf16(qa1, load_frag2(kr0 + 32, kr0 + 40), s0);
      s1 = wmma_bf16(qa0, load_frag2(kr1,      kr1 + 8),  s1);
      s1 = wmma_bf16(qa1, load_frag2(kr1 + 32, kr1 + 40), s1);
    }
    const float k0n0 = k0b[kv + lr];
    const float k0n1 = k0b[kv + 16 + lr];

    float sc0[8], sc1[8];
    #pragma unroll
    for (int r = 0; r < 8; ++r) {
      const int m = m0 + h * 8 + r;
      const unsigned char* mp = mb + (size_t)m * S_LEN + kv;
      const float v0 = (s0[r] - q0r[r] * k0n0) * scale;
      const float v1 = (s1[r] - q0r[r] * k0n1) * scale;
      sc0[r] = mp[lr]      ? v0 : -1e9f;
      sc1[r] = mp[16 + lr] ? v1 : -1e9f;
    }

    // ---- online softmax; stash P (bf16) into LDS in row-major 16x32 ----
    float alpha[8];
    #pragma unroll
    for (int r = 0; r < 8; ++r) {
      const float cm = redmax16(fmaxf(sc0[r], sc1[r]));
      const float mn = fmaxf(mrow[r], cm);
      const float p0 = __expf(sc0[r] - mn);
      const float p1 = __expf(sc1[r] - mn);
      const float rs = redsum16(p0 + p1);
      alpha[r] = __expf(mrow[r] - mn);
      lrow[r] = lrow[r] * alpha[r] + rs;
      mrow[r] = mn;
      Pl[(h * 8 + r) * PLROW + lr]      = f2bf(p0);
      Pl[(h * 8 + r) * PLROW + 16 + lr] = f2bf(p1);
    }
    #pragma unroll
    for (int t = 0; t < 5; ++t)
      #pragma unroll
      for (int r = 0; r < 8; ++r)
        acc[t][r] *= alpha[r];

    // wave-private LDS region: stores and loads from the same wave; drain DS
    asm volatile("s_wait_dscnt 0x0" ::: "memory");

    // P as A-fragment (16 queries x 32 keys)
    const unsigned short* prow = Pl + lr * PLROW;
    const v16bf pa = load_frag2(prow + 16 * h, prow + 16 * h + 8);

    // P @ V over 5 N-tiles of dims (time + space + pad), K = 32 keys
    #pragma unroll
    for (int t = 0; t < 5; ++t) {
      const unsigned short* vrow = &Vl[p][(t * 16 + lr) * VLROW + 16 * h];
      acc[t] = wmma_bf16(pa, load_frag2(vrow, vrow + 8), acc[t]);
    }

    // own async copies done before the barrier; barrier also guarantees no
    // wave rewrites buf p^1 while another is still reading it
    wait_async0();
    __syncthreads();
  }

  // ---- finalize: softmax divide + Lorentz midpoint renorm ----
  float fac[8];
  #pragma unroll
  for (int r = 0; r < 8; ++r) {
    const float inv = 1.0f / lrow[r];
    float part = 0.0f;
    #pragma unroll
    for (int t = 0; t < 5; ++t) {
      const float v = acc[t][r] * inv;
      acc[t][r] = v;
      const int d = t * 16 + lr;
      const float sgn = (d == 0) ? -1.0f : 1.0f;   // <x,x>_L = ||s||^2 - t^2
      part += (d < 65) ? sgn * v * v : 0.0f;
    }
    const float inner = redsum16(part);
    fac[r] = rsqrtf(fmaxf(-inner, 1e-6f));         // 1/(sqrt(c)*lnorm)
  }

  #pragma unroll
  for (int r = 0; r < 8; ++r) {
    const int m = m0 + h * 8 + r;
    const size_t base = ((size_t)(bb * S_LEN + m) * NH + head) * 65;
    #pragma unroll
    for (int t = 0; t < 5; ++t) {
      const int d = t * 16 + lr;
      if (d < 65) mid[base + d] = acc[t][r] * fac[r];
    }
  }
}

// -------- head fusion (Lorentz midpoint) + output projection + lift --------
__global__ __launch_bounds__(512)
void k_out(const float* __restrict__ mid, const float* __restrict__ Wo,
           const float* __restrict__ bo, float* __restrict__ out) {
  __shared__ float fused[80];
  __shared__ float red[17];
  const int row = blockIdx.x;            // b*S + s
  const int tid = threadIdx.x;
  const float* mrow = mid + (size_t)row * NH * 65;

  if (tid < 65) {
    float a = 0.0f;
    #pragma unroll
    for (int hh = 0; hh < NH; ++hh) a += mrow[hh * 65 + tid];
    fused[tid] = a * (1.0f / NH);
  }
  __syncthreads();
  if (tid == 0) {
    float inner = -fused[0] * fused[0];
    for (int d = 1; d < 65; ++d) inner += fused[d] * fused[d];
    red[16] = rsqrtf(fmaxf(-inner, 1e-6f));
  }
  __syncthreads();
  const float fac = red[16];

  float o = bo[tid];
  const float* wrow = Wo + (size_t)tid * 65;
  #pragma unroll 5
  for (int k = 0; k < 65; ++k) o += fused[k] * fac * wrow[k];

  float p = o * o;
  #pragma unroll
  for (int m = 1; m < 32; m <<= 1) p += __shfl_xor(p, m, 32);
  if ((tid & 31) == 0) red[tid >> 5] = p;
  __syncthreads();
  float* orow = out + (size_t)row * DP1;
  if (tid == 0) {
    float s2 = 0.0f;
    #pragma unroll
    for (int w = 0; w < 16; ++w) s2 += red[w];
    orow[0] = sqrtf(1.0f + s2);                     // lift: t = sqrt(1/c+||s||^2)
  }
  orow[1 + tid] = o;
}

// ------------------------------- launch -------------------------------
extern "C" void kernel_launch(void* const* d_in, const int* in_sizes, int n_in,
                              void* d_out, int out_size, void* d_ws, size_t ws_size,
                              hipStream_t stream) {
  const float* x  = (const float*)d_in[0];
  const float* Wq = (const float*)d_in[1];
  const float* bq = (const float*)d_in[2];
  const float* Wk = (const float*)d_in[3];
  const float* bk = (const float*)d_in[4];
  const float* Wv = (const float*)d_in[5];
  const float* bv = (const float*)d_in[6];
  const float* Wo = (const float*)d_in[7];
  const float* bo = (const float*)d_in[8];
  const unsigned char* mask = (const unsigned char*)d_in[9];
  float* out = (float*)d_out;

  char* ws = (char*)d_ws;
  size_t off = 0;
  auto carve = [&](size_t bytes) {
    void* p = ws + off;
    off = (off + bytes + 255) & ~(size_t)255;
    return p;
  };
  unsigned short* xbf  = (unsigned short*)carve((size_t)ROWS * KPAD * 2);
  unsigned short* wqbf = (unsigned short*)carve((size_t)D_HID * KPAD * 2);
  unsigned short* wkbf = (unsigned short*)carve((size_t)D_HID * KPAD * 2);
  unsigned short* wvbf = (unsigned short*)carve((size_t)D_HID * KPAD * 2);
  unsigned short* qs   = (unsigned short*)carve((size_t)BQ * NH * S_LEN * DH * 2);
  float*          q0f  = (float*)carve((size_t)BQ * NH * S_LEN * 4);
  unsigned short* ks   = (unsigned short*)carve((size_t)BQ * NH * S_LEN * DH * 2);
  float*          k0f  = (float*)carve((size_t)BQ * NH * S_LEN * 4);
  unsigned short* vt   = (unsigned short*)carve((size_t)BQ * NH * VROWS * S_LEN * 2);
  float*          mid  = (float*)carve((size_t)BQ * S_LEN * NH * 65 * 4);

  const int nx = ROWS * KPAD;
  k_pack<<<(nx + 255) / 256, 256, 0, stream>>>(x, xbf, ROWS, DP1, KPAD);
  const int nw = D_HID * KPAD;
  k_pack<<<(nw + 255) / 256, 256, 0, stream>>>(Wq, wqbf, D_HID, DP1, KPAD);
  k_pack<<<(nw + 255) / 256, 256, 0, stream>>>(Wk, wkbf, D_HID, DP1, KPAD);
  k_pack<<<(nw + 255) / 256, 256, 0, stream>>>(Wv, wvbf, D_HID, DP1, KPAD);
  const int nv = BQ * NH * VROWS * S_LEN;
  k_zero<<<(nv + 255) / 256, 256, 0, stream>>>(vt, nv);

  dim3 gp(ROWS / (16 * 8), NH);
  k_project<<<gp, 256, 0, stream>>>(xbf, wqbf, bq, qs, q0f, nullptr, 0);
  k_project<<<gp, 256, 0, stream>>>(xbf, wkbf, bk, ks, k0f, nullptr, 1);
  k_project<<<gp, 256, 0, stream>>>(xbf, wvbf, bv, nullptr, nullptr, vt, 2);

  dim3 ga(S_LEN / 128, NH, BQ);
  k_attn<<<ga, 256, 0, stream>>>(qs, q0f, ks, k0f, vt, mask, mid);

  k_out<<<BQ * S_LEN, 512, 0, stream>>>(mid, Wo, bo, out);
}
